// GCN_Sparse_Memory_3_66726611911076
// MI455X (gfx1250) — matrix operand order repaired and checked by reading, not compiled, not run on hardware
//
#include <hip/hip_runtime.h>

#define FDIM 128

typedef __attribute__((ext_vector_type(16))) __bf16       v16bf;
typedef __attribute__((ext_vector_type(8)))  __bf16       v8bf;
typedef __attribute__((ext_vector_type(8)))  float        v8f;
typedef __attribute__((ext_vector_type(4)))  float        v4f;
typedef __attribute__((ext_vector_type(2)))  unsigned int v2u;

// ---------- bf16 helpers (RNE) ----------
static __device__ __forceinline__ unsigned short f2bfu(float f) {
  unsigned u = __builtin_bit_cast(unsigned, f);
  u += 0x7FFFu + ((u >> 16) & 1u);
  return (unsigned short)(u >> 16);
}
static __device__ __forceinline__ __bf16 f2bf(float f) {
  unsigned short h = f2bfu(f);
  return __builtin_bit_cast(__bf16, h);
}
static __device__ __forceinline__ float bflo2f(unsigned u) { // bf16 in bits [15:0]
  return __builtin_bit_cast(float, u << 16);
}
static __device__ __forceinline__ float bfhi2f(unsigned u) { // bf16 in bits [31:16]
  return __builtin_bit_cast(float, u & 0xFFFF0000u);
}

// ---------- 0a) one-time: f32 -> bf16 elementwise (4 elems/thread) ----------
__global__ __launch_bounds__(256)
void gcn_cvt_bf16(const float* __restrict__ in, __bf16* __restrict__ out,
                  int n4 /* = total/4 */) {
  int t = blockIdx.x * 256 + threadIdx.x;
  if (t >= n4) return;
  v4f x = *(const v4f*)(in + (size_t)t * 4);
  v2u pk;
  pk[0] = (unsigned)f2bfu(x[0]) | ((unsigned)f2bfu(x[1]) << 16);
  pk[1] = (unsigned)f2bfu(x[2]) | ((unsigned)f2bfu(x[3]) << 16);
  *(v2u*)(out + (size_t)t * 4) = pk;
}

// ---------- 0b) one-time: W[k][n] f32 -> Wt[n][k] bf16 (transpose+convert) ----
__global__ __launch_bounds__(256)
void gcn_prep_w(const float* __restrict__ W, __bf16* __restrict__ Wt) {
  int t = blockIdx.x * 256 + threadIdx.x;   // 16384 threads: t = n*128 + k
  int n = t >> 7, k = t & 127;
  Wt[(size_t)n * FDIM + k] = f2bf(W[(size_t)k * FDIM + n]);
}

// ---------- 1) row_ptr from sorted rows: rp[i] = lower_bound(rows, i) ----------
__global__ __launch_bounds__(256)
void gcn_build_rowptr(const int* __restrict__ rows, int* __restrict__ rp,
                      int nrows, int E) {
  int i = blockIdx.x * 256 + threadIdx.x;
  if (i > nrows) return;
  int lo = 0, hi = E;
  while (lo < hi) {
    int mid = (lo + hi) >> 1;
    if (rows[mid] < i) lo = mid + 1; else hi = mid;
  }
  rp[i] = lo;
}

// ---------- 2) GEMM: out(bf16)[M,128] = A(bf16)[M,128] @ Wt(bf16)[n][k] ------
// Block = 256 thr = 8 waves; block owns 16 rows, wave w owns N-tile w*16.
// K=128 -> 4 x v_wmma_f32_16x16x32_bf16; all fragment loads are b128.
__global__ __launch_bounds__(256)
void gcn_gemm(const __bf16* __restrict__ A, const __bf16* __restrict__ Wt,
              __bf16* __restrict__ out, int M) {
  const int lane = threadIdx.x & 31;
  const int wave = threadIdx.x >> 5;
  const int half = lane >> 4;      // 0: lanes 0-15, 1: lanes 16-31
  const int lm   = lane & 15;
  const int row0 = blockIdx.x * 16;
  const int col0 = wave * 16;

  int arow = row0 + lm;
  if (arow >= M) arow = M - 1;     // clamp (EXEC must stay all-1 for WMMA)

  const __bf16* Abase = A  + (size_t)arow * FDIM;
  const __bf16* Bbase = Wt + (size_t)(col0 + lm) * FDIM;

  v8f acc = {};
  #pragma unroll
  for (int kk = 0; kk < FDIM; kk += 32) {
    // A frag 16x32: lane holds K = kk+8*half+{0..7} and +{16..23} (2 x b128)
    const __bf16* Ar = Abase + kk + half * 8;
    v8bf alo = *(const v8bf*)(Ar + 0);
    v8bf ahi = *(const v8bf*)(Ar + 16);
    // B frag 32x16: lane = column col0+lm, K = kk+8*half+{0..7} and +{16..23} (2 x b128)
    const __bf16* Bc = Bbase + kk + half * 8;
    v8bf blo = *(const v8bf*)(Bc + 0);
    v8bf bhi = *(const v8bf*)(Bc + 16);
    v16bf a, b;
    #pragma unroll
    for (int i = 0; i < 8; ++i) {
      a[i] = alo[i]; a[8 + i] = ahi[i];
      b[i] = blo[i]; b[8 + i] = bhi[i];
    }
    acc = __builtin_amdgcn_wmma_f32_16x16x32_bf16(
        /*neg_a=*/false, a, /*neg_b=*/false, b,
        /*c_mod=*/(short)0, acc, /*reuse_a=*/false, /*reuse_b=*/false);
  }
  // D layout: VGPR j -> M = row0 + j + 8*half, N = col0 + lm.
  // Uniform bounds branch: full tiles store unguarded (single clause).
  __bf16* orow = out + (size_t)(row0 + half * 8) * FDIM + col0 + lm;
  if (row0 + 16 <= M) {
    #pragma unroll
    for (int j = 0; j < 8; ++j) orow[(size_t)j * FDIM] = f2bf(acc[j]);
  } else {
    #pragma unroll
    for (int j = 0; j < 8; ++j)
      if (row0 + half * 8 + j < M) orow[(size_t)j * FDIM] = f2bf(acc[j]);
  }
}

// ---------- 3) SpMM: out[row] = (relu)(sum_e vals[e]*X[cols[e]] + bias) ------
// One wave per row (CSR via rp). Lane owns 4 feature cols; gathers are
// uniform-base (SGPR col index) + lane-offset b64 loads of bf16x4.
static __device__ __forceinline__ void fma_bf4(v4f& acc, v2u p, float w) {
  acc[0] = __builtin_fmaf(w, bflo2f(p[0]), acc[0]);
  acc[1] = __builtin_fmaf(w, bfhi2f(p[0]), acc[1]);
  acc[2] = __builtin_fmaf(w, bflo2f(p[1]), acc[2]);
  acc[3] = __builtin_fmaf(w, bfhi2f(p[1]), acc[3]);
}

template <bool OUT_BF16, bool RELU>
__global__ __launch_bounds__(256)
void gcn_spmm(const __bf16* __restrict__ Xb, const int* __restrict__ rp,
              const int* __restrict__ cols, const float* __restrict__ vals,
              const float* __restrict__ bias, void* __restrict__ outv,
              int nrows) {
  const int lane = threadIdx.x & 31;
  const int wave = __builtin_amdgcn_readfirstlane((int)(threadIdx.x >> 5));
  const int row  = blockIdx.x * 8 + wave;
  if (row >= nrows) return;

  int s = __builtin_amdgcn_readfirstlane(rp[row]);
  int e = __builtin_amdgcn_readfirstlane(rp[row + 1]);
  const int fo = lane * 4;   // 4 bf16 feature elements per lane (8 bytes)

  v4f acc = {};
  int i = s;
  for (; i + 4 <= e; i += 4) {            // 4-edge unroll: loads in flight
    int c0 = __builtin_amdgcn_readfirstlane(cols[i + 0]);
    int c1 = __builtin_amdgcn_readfirstlane(cols[i + 1]);
    int c2 = __builtin_amdgcn_readfirstlane(cols[i + 2]);
    int c3 = __builtin_amdgcn_readfirstlane(cols[i + 3]);
    float w0 = vals[i + 0], w1 = vals[i + 1];
    float w2 = vals[i + 2], w3 = vals[i + 3];
    v2u p0 = *(const v2u*)(Xb + (size_t)c0 * FDIM + fo);
    v2u p1 = *(const v2u*)(Xb + (size_t)c1 * FDIM + fo);
    v2u p2 = *(const v2u*)(Xb + (size_t)c2 * FDIM + fo);
    v2u p3 = *(const v2u*)(Xb + (size_t)c3 * FDIM + fo);
    fma_bf4(acc, p0, w0);
    fma_bf4(acc, p1, w1);
    fma_bf4(acc, p2, w2);
    fma_bf4(acc, p3, w3);
  }
  for (; i < e; ++i) {
    int c = __builtin_amdgcn_readfirstlane(cols[i]);
    float w = vals[i];
    v2u p = *(const v2u*)(Xb + (size_t)c * FDIM + fo);
    fma_bf4(acc, p, w);
  }

  v4f bb = *(const v4f*)(bias + fo);
  v4f r  = acc + bb;
  if (RELU) {
    #pragma unroll
    for (int j = 0; j < 4; ++j) r[j] = fmaxf(r[j], 0.0f);
  }
  if (OUT_BF16) {
    __bf16* out = (__bf16*)outv;
    v2u pk;
    pk[0] = (unsigned)f2bfu(r[0]) | ((unsigned)f2bfu(r[1]) << 16);
    pk[1] = (unsigned)f2bfu(r[2]) | ((unsigned)f2bfu(r[3]) << 16);
    *(v2u*)(out + (size_t)row * FDIM + fo) = pk;
  } else {
    float* out = (float*)outv;
    *(v4f*)(out + (size_t)row * FDIM + fo) = r;
  }
}

// ---------- launcher ----------
extern "C" void kernel_launch(void* const* d_in, const int* in_sizes, int n_in,
                              void* d_out, int out_size, void* d_ws, size_t ws_size,
                              hipStream_t stream) {
  const float* features = (const float*)d_in[0];
  const int*   adj_rows = (const int*)d_in[1];
  const int*   adj_cols = (const int*)d_in[2];
  const float* adj_vals = (const float*)d_in[3];
  const float* W1       = (const float*)d_in[4];
  const float* b1       = (const float*)d_in[5];
  const float* W2       = (const float*)d_in[6];
  const float* b2       = (const float*)d_in[7];
  (void)n_in; (void)out_size; (void)ws_size;

  const int N = in_sizes[0] / FDIM;   // 100000 nodes
  const int E = in_sizes[1];          // 3.2M edges

  // workspace layout (256B aligned):
  // [rp int(N+1)] [Xbf bf16 N*128] [bufA bf16 N*128] [bufB bf16 N*128]
  // [W1t bf16 128*128] [W2t bf16 128*128]                      (~77 MB)
  char*  ws   = (char*)d_ws;
  size_t o    = 0;
  int*    rp   = (int*)(ws + o);    o += (((size_t)(N + 1) * 4) + 255) & ~(size_t)255;
  size_t  fb   = (((size_t)N * FDIM * 2) + 255) & ~(size_t)255;
  __bf16* Xbf  = (__bf16*)(ws + o); o += fb;
  __bf16* bufA = (__bf16*)(ws + o); o += fb;
  __bf16* bufB = (__bf16*)(ws + o); o += fb;
  __bf16* W1t  = (__bf16*)(ws + o); o += FDIM * FDIM * 2;
  __bf16* W2t  = (__bf16*)(ws + o);

  // one-time prep (streaming; negligible vs SpMM)
  gcn_build_rowptr<<<(N + 1 + 255) / 256, 256, 0, stream>>>(adj_rows, rp, N, E);
  gcn_cvt_bf16<<<((N * FDIM / 4) + 255) / 256, 256, 0, stream>>>(features, Xbf,
                                                                 N * FDIM / 4);
  gcn_prep_w<<<(FDIM * FDIM) / 256, 256, 0, stream>>>(W1, W1t);
  gcn_prep_w<<<(FDIM * FDIM) / 256, 256, 0, stream>>>(W2, W2t);

  // Layer 1: H = X @ W1 ; B1 = relu(A·H + b1)
  gcn_gemm<<<(N + 15) / 16, 256, 0, stream>>>(Xbf, W1t, bufA, N);
  gcn_spmm<true,  true ><<<(N + 7) / 8, 256, 0, stream>>>(bufA, rp, adj_cols,
                                                          adj_vals, b1, bufB, N);
  // Layer 2: G = B1 @ W2 ; out = A·G + b2   (fp32 output)
  gcn_gemm<<<(N + 15) / 16, 256, 0, stream>>>(bufB, W2t, bufA, N);
  gcn_spmm<false, false><<<(N + 7) / 8, 256, 0, stream>>>(bufA, rp, adj_cols,
                                                          adj_vals, b2, d_out, N);
}
